// LSTMDecoder_55705725829741
// MI455X (gfx1250) — matrix operand order repaired and checked by reading, not compiled, not run on hardware
//
#include <hip/hip_runtime.h>
#include <hip/hip_bf16.h>
#include <math.h>

// ---------------- problem constants ----------------
#define BATCH 64
#define HID   1024
#define EMB   1024
#define VOC   32000
#define TT    50
#define G4H   (4 * HID)      // 4096
#define START_INDEX 1

// ---------------- WMMA types ----------------
typedef __bf16 v16bf __attribute__((ext_vector_type(16)));
typedef float  v8f   __attribute__((ext_vector_type(8)));

struct u4raw { unsigned x, y, z, w; };

// Load a 16-element bf16 fragment from two 16B chunks (global or LDS pointer).
__device__ __forceinline__ v16bf load16bf(const unsigned short* p0,
                                          const unsigned short* p1) {
    v16bf r;
    ((u4raw*)&r)[0] = *(const u4raw*)p0;   // elements 0..7
    ((u4raw*)&r)[1] = *(const u4raw*)p1;   // elements 8..15
    return r;
}

// A fragment (row-major M x K source, 16x32 tile):
// lanes 0-15 hold K = k0+0..7 and k0+16..23 of row m0+lane
// lanes 16-31 hold K = k0+8..15 and k0+24..31 of row m0+lane-16
__device__ __forceinline__ v16bf load_a_frag(const unsigned short* A, int lda,
                                             int row, int k0, int lhalf) {
    const unsigned short* base = A + row * lda + k0 + lhalf * 8;
    return load16bf(base, base + 16);
}

// B fragment from row-major (N x K) weights (i.e. computing X @ W^T):
// lane (0-15) holds column n0+lane, K = k0 + lhalf*8 .. +7 and k0 + 16 + lhalf*8 .. +7.
__device__ __forceinline__ v16bf load_b_frag(const unsigned short* W, int ldb,
                                             int col, int k0, int lhalf) {
    const unsigned short* base = W + col * ldb + k0 + lhalf * 8;
    return load16bf(base, base + 16);
}

__device__ __forceinline__ unsigned short f2bf(float x) {
    unsigned u = __float_as_uint(x);
    unsigned r = (u + 0x7FFFu + ((u >> 16) & 1u)) >> 16;   // RNE
    return (unsigned short)r;
}

// Async copy of one 16B chunk per lane: global -> LDS (ASYNCcnt-tracked).
__device__ __forceinline__ void async_g2l_b128(unsigned lds_byte_addr,
                                               const unsigned short* gaddr) {
    asm volatile("global_load_async_to_lds_b128 %0, %1, off"
                 :: "v"(lds_byte_addr), "v"(gaddr) : "memory");
}
__device__ __forceinline__ void wait_async0() {
    asm volatile("s_wait_asynccnt 0x0" ::: "memory");
}

// ---------------- prep kernels ----------------
__global__ void cvt_f32_bf16(const float* __restrict__ in,
                             unsigned short* __restrict__ out, int n) {
    int i = blockIdx.x * blockDim.x + threadIdx.x;
    int stride = gridDim.x * blockDim.x;
    for (; i < n; i += stride) out[i] = f2bf(in[i]);
}

__global__ void init_state(const float* __restrict__ hidden,
                           float* __restrict__ c,
                           unsigned short* __restrict__ hB,
                           int* __restrict__ pred) {
    int i = blockIdx.x * blockDim.x + threadIdx.x;   // 65536 threads
    if (i < BATCH * HID) {
        c[i] = 0.0f;
        hB[i] = f2bf(hidden[i]);
    }
    if (i < BATCH) pred[i] = START_INDEX;
}

// x_bf16[b][:] = bf16(embedding[pred[b]][:])
__global__ void embed_gather(const float* __restrict__ embedding,
                             const int* __restrict__ pred,
                             unsigned short* __restrict__ xB) {
    int b = blockIdx.x;
    int idx = pred[b];
    const float* src = embedding + (size_t)idx * EMB;
    for (int j = threadIdx.x; j < EMB; j += blockDim.x)
        xB[b * EMB + j] = f2bf(src[j]);
}

// ---------------- gates GEMM: (64x1024)x(1024x4096) twice, bf16 WMMA --------
// x and h (2 x 128 KB) async-staged to LDS; one wave per 16-wide N strip over
// all of M=64; W_ih/W_hh fragment streams double-buffered so WMMAs never wait
// on a just-issued global load. grid 32 blocks x 256 threads (8 waves).
__global__ void __launch_bounds__(256)
gates_gemm(const unsigned short* __restrict__ xB,
           const unsigned short* __restrict__ hB,
           const unsigned short* __restrict__ WihB,
           const unsigned short* __restrict__ WhhB,
           float* __restrict__ gates) {
    extern __shared__ __align__(16) unsigned short smem[];  // [0,64K): x  [64K,128K): h (ushort idx)
    const int K = HID, N = G4H;
    const int CH = (BATCH * HID) / 8;                       // 8192 b128 chunks per matrix
    int tid = threadIdx.x;

    for (int i = tid; i < 2 * CH; i += 256) {               // boundary 8192 is 256-aligned: uniform
        const unsigned short* g = (i < CH) ? (xB + i * 8) : (hB + (i - CH) * 8);
        async_g2l_b128((unsigned)(size_t)&smem[i * 8], g);
    }
    wait_async0();
    __syncthreads();
    const unsigned short* sX = smem;
    const unsigned short* sH = smem + BATCH * HID;

    int wave = tid >> 5, lane = tid & 31;
    int lmod = lane & 15, lhalf = lane >> 4;
    int n0 = (blockIdx.x * 8 + wave) * 16;                  // 0..4080

    v8f acc[4] = {{}, {}, {}, {}};
    v16bf bi = load_b_frag(WihB, K, n0 + lmod, 0, lhalf);
    v16bf bh = load_b_frag(WhhB, K, n0 + lmod, 0, lhalf);
    for (int k0 = 0; k0 < K - 32; k0 += 32) {
        v16bf bi_n = load_b_frag(WihB, K, n0 + lmod, k0 + 32, lhalf);
        v16bf bh_n = load_b_frag(WhhB, K, n0 + lmod, k0 + 32, lhalf);
        #pragma unroll
        for (int mb = 0; mb < 4; ++mb) {
            v16bf a = load_a_frag(sX, K, mb * 16 + lmod, k0, lhalf);  // ds_load
            acc[mb] = __builtin_amdgcn_wmma_f32_16x16x32_bf16(
                false, a, false, bi, (short)0, acc[mb], false, false);
        }
        #pragma unroll
        for (int mb = 0; mb < 4; ++mb) {
            v16bf a = load_a_frag(sH, K, mb * 16 + lmod, k0, lhalf);  // ds_load
            acc[mb] = __builtin_amdgcn_wmma_f32_16x16x32_bf16(
                false, a, false, bh, (short)0, acc[mb], false, false);
        }
        bi = bi_n; bh = bh_n;
    }
    {   // final k-block (K-32)
        const int k0 = K - 32;
        #pragma unroll
        for (int mb = 0; mb < 4; ++mb) {
            v16bf a = load_a_frag(sX, K, mb * 16 + lmod, k0, lhalf);
            acc[mb] = __builtin_amdgcn_wmma_f32_16x16x32_bf16(
                false, a, false, bi, (short)0, acc[mb], false, false);
        }
        #pragma unroll
        for (int mb = 0; mb < 4; ++mb) {
            v16bf a = load_a_frag(sH, K, mb * 16 + lmod, k0, lhalf);
            acc[mb] = __builtin_amdgcn_wmma_f32_16x16x32_bf16(
                false, a, false, bh, (short)0, acc[mb], false, false);
        }
    }
    int n = n0 + lmod;
    #pragma unroll
    for (int mb = 0; mb < 4; ++mb) {
        #pragma unroll
        for (int r = 0; r < 8; ++r) {
            int m = mb * 16 + r + 8 * lhalf;
            gates[m * N + n] = acc[mb][r];
        }
    }
}

// ---------------- LSTM pointwise ----------------
__global__ void lstm_pointwise(const float* __restrict__ gates,
                               const float* __restrict__ b_ih,
                               const float* __restrict__ b_hh,
                               float* __restrict__ c,
                               unsigned short* __restrict__ hB) {
    int idx = blockIdx.x * blockDim.x + threadIdx.x;   // 65536
    int b = idx >> 10, j = idx & 1023;
    const float* g = gates + b * G4H;
    float gi = g[j]        + b_ih[j]        + b_hh[j];
    float gf = g[j + HID]  + b_ih[j + HID]  + b_hh[j + HID];
    float gg = g[j + 2*HID]+ b_ih[j + 2*HID]+ b_hh[j + 2*HID];
    float go = g[j + 3*HID]+ b_ih[j + 3*HID]+ b_hh[j + 3*HID];
    float si = 1.0f / (1.0f + __expf(-gi));
    float sf = 1.0f / (1.0f + __expf(-gf));
    float so = 1.0f / (1.0f + __expf(-go));
    float cn = sf * c[idx] + si * tanhf(gg);
    float hn = so * tanhf(cn);
    c[idx]  = cn;
    hB[idx] = f2bf(hn);
}

// ---------------- logits GEMM: (64x1024)x(1024x32000) + b_out, bf16 WMMA ----
// h (128 KB) async-staged to LDS once per block; one wave per 16-wide N strip
// over all of M=64 (weight stream read exactly once per step, L2-resident);
// W_out fragment stream double-buffered. grid 250 blocks x 256 threads.
__global__ void __launch_bounds__(256)
logits_gemm(const unsigned short* __restrict__ hB,
            const unsigned short* __restrict__ WoutB,
            const float* __restrict__ b_out,
            float* __restrict__ logits) {
    extern __shared__ __align__(16) unsigned short sA[];    // 64 x 1024 bf16 = 128 KB
    const int K = HID, N = VOC;
    const int CH = (BATCH * HID) / 8;                       // 8192 b128 chunks
    int tid = threadIdx.x;

    for (int i = tid; i < CH; i += 256)
        async_g2l_b128((unsigned)(size_t)&sA[i * 8], hB + i * 8);
    wait_async0();
    __syncthreads();

    int wave = tid >> 5, lane = tid & 31;
    int lmod = lane & 15, lhalf = lane >> 4;
    int n0 = (blockIdx.x * 8 + wave) * 16;                  // 0..31984

    v8f acc[4] = {{}, {}, {}, {}};
    v16bf bcur = load_b_frag(WoutB, K, n0 + lmod, 0, lhalf);
    for (int k0 = 0; k0 < K - 32; k0 += 32) {
        v16bf bnxt = load_b_frag(WoutB, K, n0 + lmod, k0 + 32, lhalf);
        #pragma unroll
        for (int mb = 0; mb < 4; ++mb) {
            v16bf a = load_a_frag(sA, K, mb * 16 + lmod, k0, lhalf);  // ds_load
            acc[mb] = __builtin_amdgcn_wmma_f32_16x16x32_bf16(
                false, a, false, bcur, (short)0, acc[mb], false, false);
        }
        bcur = bnxt;
    }
    {   // final k-block
        const int k0 = K - 32;
        #pragma unroll
        for (int mb = 0; mb < 4; ++mb) {
            v16bf a = load_a_frag(sA, K, mb * 16 + lmod, k0, lhalf);
            acc[mb] = __builtin_amdgcn_wmma_f32_16x16x32_bf16(
                false, a, false, bcur, (short)0, acc[mb], false, false);
        }
    }
    int n = n0 + lmod;
    float bias = b_out[n];
    #pragma unroll
    for (int mb = 0; mb < 4; ++mb) {
        #pragma unroll
        for (int r = 0; r < 8; ++r) {
            int m = mb * 16 + r + 8 * lhalf;
            logits[m * N + n] = acc[mb][r] + bias;
        }
    }
}

// ---------------- softmax + argmax (one block per batch row) ----------------
__global__ void softmax_argmax(const float* __restrict__ logits,
                               float* __restrict__ probs_out,   // (B, T, V)
                               float* __restrict__ preds_out,   // (B, T) as float
                               int* __restrict__ pred_ws, int t) {
    __shared__ float smax[256];
    __shared__ int   sarg[256];
    __shared__ float ssum[256];
    int b = blockIdx.x, tid = threadIdx.x;
    const float* row = logits + (size_t)b * VOC;

    float mx = -3.4e38f; int arg = 0;
    for (int v = tid; v < VOC; v += 256) {
        float x = row[v];
        if (x > mx) { mx = x; arg = v; }   // strict > keeps first index per thread
    }
    smax[tid] = mx; sarg[tid] = arg;
    __syncthreads();
    for (int s = 128; s > 0; s >>= 1) {
        if (tid < s) {
            float a = smax[tid], bb = smax[tid + s];
            if (bb > a || (bb == a && sarg[tid + s] < sarg[tid])) {
                smax[tid] = bb; sarg[tid] = sarg[tid + s];
            }
        }
        __syncthreads();
    }
    float rowmax = smax[0];
    int   rowarg = sarg[0];

    float sum = 0.0f;
    for (int v = tid; v < VOC; v += 256) sum += __expf(row[v] - rowmax);
    ssum[tid] = sum;
    __syncthreads();
    for (int s = 128; s > 0; s >>= 1) {
        if (tid < s) ssum[tid] += ssum[tid + s];
        __syncthreads();
    }
    float inv = 1.0f / ssum[0];

    float* out = probs_out + ((size_t)b * TT + t) * VOC;
    for (int v = tid; v < VOC; v += 256) out[v] = __expf(row[v] - rowmax) * inv;
    if (tid == 0) {
        pred_ws[b] = rowarg;
        preds_out[b * TT + t] = (float)rowarg;
    }
}

// ---------------- orchestration ----------------
extern "C" void kernel_launch(void* const* d_in, const int* in_sizes, int n_in,
                              void* d_out, int out_size, void* d_ws, size_t ws_size,
                              hipStream_t stream) {
    const float* hidden    = (const float*)d_in[0];
    const float* embedding = (const float*)d_in[1];
    const float* W_ih      = (const float*)d_in[2];
    const float* W_hh      = (const float*)d_in[3];
    const float* b_ih      = (const float*)d_in[4];
    const float* b_hh      = (const float*)d_in[5];
    const float* W_out     = (const float*)d_in[6];
    const float* b_out     = (const float*)d_in[7];

    char* ws = (char*)d_ws;
    size_t off = 0;
    auto alloc = [&](size_t bytes) {
        void* p = ws + off;
        off = (off + bytes + 255) & ~(size_t)255;
        return p;
    };
    unsigned short* WihB  = (unsigned short*)alloc((size_t)G4H * EMB * 2);   // 8 MB
    unsigned short* WhhB  = (unsigned short*)alloc((size_t)G4H * HID * 2);   // 8 MB
    unsigned short* WoutB = (unsigned short*)alloc((size_t)VOC * HID * 2);   // 64 MB
    float*          gates = (float*)alloc((size_t)BATCH * G4H * 4);          // 1 MB
    float*          logit = (float*)alloc((size_t)BATCH * VOC * 4);          // 8 MB
    float*          cbuf  = (float*)alloc((size_t)BATCH * HID * 4);
    unsigned short* hB    = (unsigned short*)alloc((size_t)BATCH * HID * 2);
    unsigned short* xB    = (unsigned short*)alloc((size_t)BATCH * EMB * 2);
    int*            pred  = (int*)alloc(BATCH * 4);

    float* probs_out = (float*)d_out;                              // (B,T,V)
    float* preds_out = probs_out + (size_t)BATCH * TT * VOC;       // (B,T)

    // One-time (per launch) bf16 weight staging — weights then live in L2 (~80 MB of 192 MB).
    cvt_f32_bf16<<<2048, 256, 0, stream>>>(W_ih,  WihB,  G4H * EMB);
    cvt_f32_bf16<<<2048, 256, 0, stream>>>(W_hh,  WhhB,  G4H * HID);
    cvt_f32_bf16<<<8192, 256, 0, stream>>>(W_out, WoutB, VOC * HID);
    init_state<<<(BATCH * HID + 255) / 256, 256, 0, stream>>>(hidden, cbuf, hB, pred);

    const size_t ldsGates  = (size_t)2 * BATCH * HID * sizeof(unsigned short); // 256 KB
    const size_t ldsLogits = (size_t)BATCH * HID * sizeof(unsigned short);     // 128 KB

    for (int t = 0; t < TT; ++t) {
        embed_gather<<<BATCH, 256, 0, stream>>>(embedding, pred, xB);
        gates_gemm<<<32, 256, ldsGates, stream>>>(xB, hB, WihB, WhhB, gates);
        lstm_pointwise<<<(BATCH * HID + 255) / 256, 256, 0, stream>>>(
            gates, b_ih, b_hh, cbuf, hB);
        logits_gemm<<<250, 256, ldsLogits, stream>>>(hB, WoutB, b_out, logit);
        softmax_argmax<<<BATCH, 256, 0, stream>>>(logit, probs_out, preds_out, pred, t);
    }
}
